// DPPAttention_11304353923741
// MI455X (gfx1250) — matrix operand order repaired and checked by reading, not compiled, not run on hardware
//
#include <hip/hip_runtime.h>

// ---------- WMMA types (gfx1250, wave32) ----------
typedef __attribute__((ext_vector_type(16))) __bf16 v16bf;
typedef __attribute__((ext_vector_type(8)))  float  v8f;
typedef __attribute__((ext_vector_type(4)))  int    v4i;

#define BM 128
#define BN 128
#define BK 32
#define APAD 40   // padded LDS row (bf16 elems): 80B stride spreads banks
#define BPAD 40

// ---------- async global->LDS path (CDNA5 ASYNCcnt engine), guarded ----------
#if defined(__gfx1250__) && \
    __has_builtin(__builtin_amdgcn_global_load_async_to_lds_b128) && \
    __has_builtin(__builtin_amdgcn_s_wait_asynccnt)
#define USE_ASYNC_CP 1
#else
#define USE_ASYNC_CP 0
#if defined(__gfx1250__)
#warning "gfx1250: async load-to-LDS builtins not found; using sync staging"
#endif
#endif

#if USE_ASYNC_CP
typedef __attribute__((address_space(1))) v4i gv4i;  // global int4
typedef __attribute__((address_space(3))) v4i lv4i;  // LDS int4
__device__ __forceinline__ void cp16_async(__bf16* lds_dst, const __bf16* gsrc) {
  __builtin_amdgcn_global_load_async_to_lds_b128(
      (gv4i*)gsrc, (lv4i*)lds_dst, 0, 0);
}
#endif

// ---------- fp32 -> bf16 convert ----------
__global__ __launch_bounds__(256) void f2bf_k(const float* __restrict__ in,
                                              __bf16* __restrict__ out, long n) {
  long i = (long)blockIdx.x * blockDim.x + threadIdx.x;
  if (i < n) out[i] = (__bf16)in[i];
}

// ---------- per-batch transpose: xT[d][m] = x[m][d] (fp32 in -> bf16 out) ----------
__global__ __launch_bounds__(256) void transpose_bf16_k(const float* __restrict__ in,
                                                        __bf16* __restrict__ out,
                                                        int L, int D) {
  __shared__ float tile[32][33];
  int b = blockIdx.z;
  int m0 = blockIdx.y * 32, d0 = blockIdx.x * 32;
  int tx = threadIdx.x & 31, ty = threadIdx.x >> 5;   // 8 rows/pass
  const float* src = in + (long)b * L * D;
  for (int r = ty; r < 32; r += 8) tile[r][tx] = src[(long)(m0 + r) * D + d0 + tx];
  __syncthreads();
  __bf16* dst = out + (long)b * L * D;
  for (int r = ty; r < 32; r += 8)
    dst[(long)(d0 + r) * L + m0 + tx] = (__bf16)tile[tx][r];
}

// ---------- block reduction helpers (blockDim.x == 256) ----------
__device__ inline float bsum256(float* red, float v) {
  int t = threadIdx.x;
  red[t] = v; __syncthreads();
  for (int s = 128; s > 0; s >>= 1) {
    if (t < s) red[t] += red[t + s];
    __syncthreads();
  }
  float r = red[0];
  __syncthreads();
  return r;
}
__device__ inline float bmax256(float* red, float v) {
  int t = threadIdx.x;
  red[t] = v; __syncthreads();
  for (int s = 128; s > 0; s >>= 1) {
    if (t < s) red[t] = fmaxf(red[t], red[t + s]);
    __syncthreads();
  }
  float r = red[0];
  __syncthreads();
  return r;
}

// ---------- WMMA GEMM: C[M,N] = A[M,K] * B[N,K]^T ----------
// 128x128x32 block tiles, 256 threads / 8 waves, each wave a 64x32 sub-tile
// (4x2 accumulators -> 8 v_wmma per K-step vs 6 fragment loads).
// Double-buffered LDS; async global->LDS prefetch (ASYNCcnt) when available.
// EPI 0: Cf = acc                      (fp32, gram K)
// EPI 1: Cb = (acc + bias[n])^2        (bf16, squared query)
// EPI 2: Cb = acc                      (bf16, ctx)
// EPI 3: Cf = acc + bias[n] + resid    (fp32, h with residual)
template <int EPI>
__global__ __launch_bounds__(256) void gemm_bt_k(
    const __bf16* __restrict__ A, long sA,
    const __bf16* __restrict__ B, long sB,
    float* __restrict__ Cf, __bf16* __restrict__ Cb, long sC,
    const float* __restrict__ bias, const float* __restrict__ resid,
    int K, int lda, int ldb, int ldc) {
  __shared__ __bf16 As[2][BM * APAD];
  __shared__ __bf16 Bs[2][BN * BPAD];

  const int tid  = threadIdx.x;
  const int lane = tid & 31;
  const int wave = tid >> 5;
  const int lo = lane & 15;
  const int hi = lane >> 4;   // 0/1
  const int wm = wave & 1;    // 64-row band
  const int wn = wave >> 1;   // 32-col band

  const int z = blockIdx.z;
  A += (long)z * sA;
  B += (long)z * sB;
  const long coff = (long)z * sC;

  const int bm0 = blockIdx.y * BM;
  const int bn0 = blockIdx.x * BN;

  // per-thread staging addresses: thread t owns row (t>>1), 16-elem half (t&1)
  const int srow = tid >> 1;
  const int shalf = (tid & 1) * 16;
  const __bf16* gA = A + (long)(bm0 + srow) * lda + shalf;
  const __bf16* gB = B + (long)(bn0 + srow) * ldb + shalf;
  const int aoff = srow * APAD + shalf;
  const int boff = srow * BPAD + shalf;

  auto stage = [&](int buf, int k0) {
#if USE_ASYNC_CP
    cp16_async(&As[buf][aoff],     gA + k0);
    cp16_async(&As[buf][aoff + 8], gA + k0 + 8);
    cp16_async(&Bs[buf][boff],     gB + k0);
    cp16_async(&Bs[buf][boff + 8], gB + k0 + 8);
#else
    const uint4* a4 = (const uint4*)(gA + k0);
    const uint4* b4 = (const uint4*)(gB + k0);
    uint4 a0 = a4[0], a1 = a4[1];
    uint4 b0 = b4[0], b1 = b4[1];
    uint4* ad = (uint4*)(&As[buf][aoff]);
    uint4* bd = (uint4*)(&Bs[buf][boff]);
    ad[0] = a0; ad[1] = a1;
    bd[0] = b0; bd[1] = b1;
#endif
  };

  v8f acc[4][2];
  {
    v8f zero = {};
    for (int i = 0; i < 4; i++)
      for (int j = 0; j < 2; j++) acc[i][j] = zero;
  }

  const int nk = K / BK;
  stage(0, 0);
  for (int it = 0; it < nk; ++it) {
    const int buf = it & 1;
    const bool pf = (it + 1 < nk);
    if (pf) stage(buf ^ 1, (it + 1) * BK);        // prefetch next tile
#if USE_ASYNC_CP
    if (pf) __builtin_amdgcn_s_wait_asynccnt(4);  // current tile done, prefetch in flight
    else    __builtin_amdgcn_s_wait_asynccnt(0);
#endif
    __syncthreads();

    // --- fragments: contiguous k order (k = 16*hi + e) in both A and B ---
    v16bf bfrag[2];
    for (int j = 0; j < 2; j++) {
      const __bf16* bp = &Bs[buf][(wn * 32 + j * 16 + lo) * BPAD + hi * 16];
      v16bf t;
      for (int e = 0; e < 16; e++) t[e] = bp[e];   // contiguous 32B K-run
      bfrag[j] = t;
    }
    v16bf afrag[4];
    for (int i = 0; i < 4; i++) {
      const __bf16* ap = &As[buf][(wm * 64 + i * 16 + lo) * APAD];
      v16bf t;
      for (int e = 0; e < 16; e++) {
        int kk = e + hi * 16;                      // contiguous 32B K-run
        t[e] = ap[kk];
      }
      afrag[i] = t;
    }
    for (int i = 0; i < 4; i++)
      for (int j = 0; j < 2; j++)
        acc[i][j] = __builtin_amdgcn_wmma_f32_16x16x32_bf16(
            false, afrag[i], false, bfrag[j], (short)0, acc[i][j], false, false);
    __syncthreads();   // all waves done reading buf before it is re-staged
  }

  // --- epilogue: C/D layout row = r + 8*hi, col = lane%16 ---
  for (int i = 0; i < 4; i++)
    for (int j = 0; j < 2; j++) {
      int mbase = bm0 + wm * 64 + i * 16 + hi * 8;
      int nidx  = bn0 + wn * 32 + j * 16 + lo;
      for (int r = 0; r < 8; r++) {
        int row = mbase + r;
        float v = acc[i][j][r];
        long idx = coff + (long)row * ldc + nidx;
        if (EPI == 0) {
          Cf[idx] = v;
        } else if (EPI == 1) {
          float t = v + bias[nidx];
          Cb[idx] = (__bf16)(t * t);
        } else if (EPI == 2) {
          Cb[idx] = (__bf16)v;
        } else {
          Cf[idx] = v + bias[nidx] + resid[idx];
        }
      }
    }
}

// ---------- per-batch stats: diag d_i, denom via 0.5*(S1^2 - ||K||_F^2 - diag) ----------
__global__ __launch_bounds__(256) void stats_k(const float* __restrict__ Kf,
                                               float* __restrict__ dbuf,
                                               float* __restrict__ denom, int L) {
  __shared__ float red[256];
  int b = blockIdx.x;
  const float* Kb = Kf + (long)b * L * L;
  float s0 = 0.f;
  for (int i = threadIdx.x; i < L; i += 256) {
    float di = Kb[(long)i * L + i];
    dbuf[(long)b * L + i] = di;
    s0 += di;
  }
  float f = 0.f;
  long tot = (long)L * L;
  for (long t = threadIdx.x; t < tot; t += 256) {
    __builtin_prefetch(&Kb[t + 8192], 0, 0);
    float v = Kb[t];
    f += v * v;
  }
  float S0 = bsum256(red, s0);
  float F  = bsum256(red, f);
  if (threadIdx.x == 0) {
    const float e = 1e-5f;
    float S1 = S0 + (float)L * e;
    float diagsum = 2.f * e * S0 + (float)L * e * e;
    float dn = 0.5f * (S1 * S1 - F - diagsum);
    denom[b] = fmaxf(dn, 1e-9f);
  }
}

// ---------- fused det/scores/softmax; one block per (b,row); bf16 probs ----------
__global__ __launch_bounds__(256) void softmax_k(const float* __restrict__ Kf,
                                                 const float* __restrict__ dbuf,
                                                 const float* __restrict__ denom,
                                                 const float* __restrict__ mask,
                                                 __bf16* __restrict__ probs, int L) {
  __shared__ float red[256];
  int row = blockIdx.x;
  int b   = blockIdx.y;
  const long base = ((long)b * L + row) * L;
  const float* Kr = Kf + base;
  const float* mr = mask + base;
  const float* db = dbuf + (long)b * L;
  const float e = 1e-5f;
  float di = db[row];
  float inv_dn = 1.f / denom[b];

  float sc[2];
  float mx = -3.4e38f;
  for (int t = 0; t < 2; t++) {
    int j = threadIdx.x + t * 256;
    float kij = Kr[j];
    float dj  = db[j];
    float det = (di + e) * (dj + e) - kij * kij;
    float s = -(det * inv_dn + (j == row ? di : 0.f)) * 0.125f + mr[j];
    sc[t] = s;
    mx = fmaxf(mx, s);
  }
  float bmax = bmax256(red, mx);
  float ex[2];
  float sum = 0.f;
  for (int t = 0; t < 2; t++) {
    ex[t] = __expf(sc[t] - bmax);
    sum += ex[t];
  }
  float tot = bsum256(red, sum);
  float inv = 1.f / tot;
  for (int t = 0; t < 2; t++) {
    int j = threadIdx.x + t * 256;
    probs[base + j] = (__bf16)(ex[t] * inv);
  }
}

// ---------- BERT LayerNorm, one block per row ----------
__global__ __launch_bounds__(256) void ln_k(const float* __restrict__ h,
                                            const float* __restrict__ w,
                                            const float* __restrict__ bvec,
                                            float* __restrict__ out, int D) {
  __shared__ float red[256];
  long row = blockIdx.x;
  const float* hr = h + row * D;
  float s = 0.f;
  for (int i = threadIdx.x; i < D; i += 256) s += hr[i];
  float u = bsum256(red, s) / (float)D;
  float v = 0.f;
  for (int i = threadIdx.x; i < D; i += 256) {
    float t = hr[i] - u;
    v += t * t;
  }
  float var = bsum256(red, v) / (float)D;
  float inv = 1.f / sqrtf(var + 1e-12f);
  for (int i = threadIdx.x; i < D; i += 256)
    out[row * D + i] = w[i] * (hr[i] - u) * inv + bvec[i];
}

extern "C" void kernel_launch(void* const* d_in, const int* in_sizes, int n_in,
                              void* d_out, int out_size, void* d_ws, size_t ws_size,
                              hipStream_t stream) {
  (void)in_sizes; (void)n_in; (void)out_size; (void)ws_size;
  const float* x    = (const float*)d_in[0];
  const float* mask = (const float*)d_in[1];
  const float* Wq   = (const float*)d_in[2];
  const float* bq   = (const float*)d_in[3];
  const float* Wd   = (const float*)d_in[4];
  const float* bd   = (const float*)d_in[5];
  const float* lnw  = (const float*)d_in[6];
  const float* lnb  = (const float*)d_in[7];
  float* out = (float*)d_out;

  const int Bn = 64, L = 512, D = 512;
  const long BLD = (long)Bn * L * D;   // == B*L*L
  const long DD  = (long)D * D;

  char* ws = (char*)d_ws;
  size_t off = 0;
  auto alloc = [&](size_t bytes) -> void* {
    void* p = ws + off;
    off += (bytes + 255) & ~(size_t)255;
    return p;
  };
  __bf16* xb   = (__bf16*)alloc(BLD * 2);   // later aliased as x^T (bf16)
  __bf16* qlb  = (__bf16*)alloc(BLD * 2);   // later aliased as probs
  float*  Kf   = (float*) alloc(BLD * 4);   // later aliased as h
  __bf16* ctxb = (__bf16*)alloc(BLD * 2);
  __bf16* Wqb  = (__bf16*)alloc(DD * 2);
  __bf16* Wdb  = (__bf16*)alloc(DD * 2);
  float*  dbuf = (float*) alloc((long)Bn * L * 4);
  float*  dnm  = (float*) alloc((size_t)Bn * 4);
  __bf16* probs = qlb;
  __bf16* xtb   = xb;     // x^T reuses xb after the q-projection consumed it
  float*  hbuf  = Kf;

  dim3 blk(256);

  // 1) bf16 copies
  f2bf_k<<<dim3((unsigned)((BLD + 255) / 256)), blk, 0, stream>>>(x, xb, BLD);
  f2bf_k<<<dim3((unsigned)((DD + 255) / 256)),  blk, 0, stream>>>(Wq, Wqb, DD);
  f2bf_k<<<dim3((unsigned)((DD + 255) / 256)),  blk, 0, stream>>>(Wd, Wdb, DD);

  // 2) ql = (x*Wq^T + bq)^2 -> bf16
  gemm_bt_k<1><<<dim3(D / BN, (Bn * L) / BM, 1), blk, 0, stream>>>(
      xb, 0, Wqb, 0, nullptr, qlb, 0, bq, nullptr, D, D, D, D);

  // 3) x^T (bf16) per batch, overwriting xb (stream-ordered after step 2)
  transpose_bf16_k<<<dim3(D / 32, L / 32, Bn), blk, 0, stream>>>(x, xtb, L, D);

  // 4) K = ql * ql^T per batch -> fp32
  gemm_bt_k<0><<<dim3(L / BN, L / BM, Bn), blk, 0, stream>>>(
      qlb, (long)L * D, qlb, (long)L * D, Kf, nullptr, (long)L * L,
      nullptr, nullptr, D, D, D, L);

  // 5) per-batch diag + denom
  stats_k<<<dim3(Bn), blk, 0, stream>>>(Kf, dbuf, dnm, L);

  // 6) fused det/scores/softmax -> bf16 probs (aliases ql buffer)
  softmax_k<<<dim3(L, Bn), blk, 0, stream>>>(Kf, dbuf, dnm, mask, probs, L);

  // 7) ctx = probs * x = probs * (x^T)^T per batch -> bf16
  gemm_bt_k<2><<<dim3(D / BN, L / BM, Bn), blk, 0, stream>>>(
      probs, (long)L * L, xtb, (long)L * D, nullptr, ctxb, (long)L * D,
      nullptr, nullptr, L, L, L, D);

  // 8) h = ctx * Wd^T + bd + x -> fp32 (aliases K buffer)
  gemm_bt_k<3><<<dim3(D / BN, (Bn * L) / BM, 1), blk, 0, stream>>>(
      ctxb, 0, Wdb, 0, hbuf, nullptr, 0, bd, x, D, D, D, D);

  // 9) LayerNorm -> out
  ln_k<<<dim3(Bn * L), blk, 0, stream>>>(hbuf, lnw, lnb, out, D);
}
